// MultiScaleSE3Simple_79748952752296
// MI455X (gfx1250) — matrix operand order repaired and checked by reading, not compiled, not run on hardware
//
#include <hip/hip_runtime.h>
#include <hip/hip_bf16.h>

// ---------------------------------------------------------------------------
// Problem constants (from reference): N nodes, E edges, H hidden, L layers.
// ---------------------------------------------------------------------------
#define GN      50000
#define GE      800000
#define GDIN    64
#define GH      128
#define GDOUT   64
#define GED     5
#define GL      3
#define GB      128

typedef __attribute__((ext_vector_type(16))) __bf16 v16bf;
typedef __attribute__((ext_vector_type(8)))  float  v8f;

// ---------------------------------------------------------------------------
// Generic zero kernel.
// ---------------------------------------------------------------------------
__global__ void zero_f32(float* __restrict__ p, int n) {
    int i = blockIdx.x * 256 + threadIdx.x;
    if (i < n) p[i] = 0.0f;
}

// ---------------------------------------------------------------------------
// Pack a row-major fp32 weight matrix [K x Nout] (leading dim = ld) into the
// bf16 B-fragment layout consumed by v_wmma_f32_16x16x32_bf16:
//   per 16-col tile c, per 32-K block kb, per lane (32), 16 contiguous bf16.
//   lanes 0-15 : col = c*16+lane,    K = kb*32 + {0..7, 16..23}  (e = 0..15)
//   lanes 16-31: col = c*16+lane-16, K = kb*32 + {8..15, 24..31}
// One thread per packed element; packed index == flat thread index.
// ---------------------------------------------------------------------------
__global__ void pack_w(const float* __restrict__ W, __bf16* __restrict__ P,
                       int K, int Nout, int ld) {
    int idx = blockIdx.x * 256 + threadIdx.x;
    if (idx >= K * Nout) return;
    int e    = idx & 15;
    int t    = idx >> 4;
    int lane = t & 31;
    t >>= 5;
    int KB = K >> 5;
    int kb = t % KB;
    int c  = t / KB;
    int col = c * 16 + (lane & 15);
    int k   = kb * 32 + ((lane & 16) ? 8 : 0) + (e & 7) + ((e & 8) ? 16 : 0);
    P[idx] = (__bf16)W[(size_t)k * ld + col];
}

// ---------------------------------------------------------------------------
// Small fp32 matmul (128x128 @ 128x128) for Wc = edge_W2 @ updW_bot.
// ---------------------------------------------------------------------------
__global__ void mm128(const float* __restrict__ A, const float* __restrict__ B,
                      float* __restrict__ C) {
    int idx = blockIdx.x * 256 + threadIdx.x;   // 16384 threads
    int i = idx >> 7, j = idx & 127;
    float s = 0.0f;
    for (int k = 0; k < 128; ++k) s += A[i * 128 + k] * B[k * 128 + j];
    C[idx] = s;
}

// vc[j] = sum_k v[k] * B[k*128 + j]
__global__ void vecmat128(const float* __restrict__ v, const float* __restrict__ B,
                          float* __restrict__ out) {
    int j = threadIdx.x;
    float s = 0.0f;
    for (int k = 0; k < 128; ++k) s += v[k] * B[k * 128 + j];
    out[j] = s;
}

// deg[dst] += 1 for every edge
__global__ void degree_k(const int* __restrict__ dst, float* __restrict__ deg, int e_) {
    int i = blockIdx.x * 256 + threadIdx.x;
    if (i < e_) atomicAdd(&deg[dst[i]], 1.0f);
}

// ---------------------------------------------------------------------------
// WMMA GEMM:  C[M,Nout] = act( A@B  [+ A2@B2]  + bias  [+ rowsc[m]*rsvec[n]] )
//   A, A2 : fp32 row-major [M,K] / [M,K2]  (converted to bf16 in-register)
//   B, B2 : pre-packed bf16 (see pack_w)
//   Cf    : optional fp32 output, Cb: optional bf16 output
// One wave computes a 16x64 C tile: one A fragment feeds four back-to-back
// v_wmma_f32_16x16x32_bf16 (independent accumulators -> no RAW hazard NOPs),
// K stepped by 32.  M % 16 == 0, Nout % 64 == 0, K % 32 == 0 (true here).
// ---------------------------------------------------------------------------
__global__ __launch_bounds__(256) void gemm_wmma(
    const float* __restrict__ A,  const __bf16* __restrict__ Bp,
    const float* __restrict__ A2, const __bf16* __restrict__ B2p,
    const float* __restrict__ bias,
    const float* __restrict__ rowsc, const float* __restrict__ rsvec,
    float* __restrict__ Cf, __bf16* __restrict__ Cb,
    int M, int K, int K2, int Nout, int relu_flag)
{
    const int wave = threadIdx.x >> 5;
    const int lane = threadIdx.x & 31;
    const int cgroups = Nout >> 6;                  // 64-wide column groups
    const int mtiles  = M >> 4;
    int tile = blockIdx.x * 8 + wave;
    if (tile >= cgroups * mtiles) return;           // wave-uniform exit
    const int tm = tile / cgroups;
    const int cg = tile % cgroups;

    const int arow = tm * 16 + (lane & 15);
    const int koff = (lane & 16) ? 8 : 0;

    v8f c0 = {0.f,0.f,0.f,0.f,0.f,0.f,0.f,0.f};
    v8f c1 = c0, c2 = c0, c3 = c0;

    const int KB = K >> 5;
    for (int kb = 0; kb < KB; ++kb) {
        // A fragment: 16-bit A-matrix 16x32 layout (ISA 7.12.2)
        const float* ap = A + (size_t)arow * K + kb * 32 + koff;
        v16bf a;
#pragma unroll
        for (int i = 0; i < 8; ++i) a[i] = (__bf16)ap[i];          // K = +0..7
#pragma unroll
        for (int i = 0; i < 8; ++i) a[8 + i] = (__bf16)ap[16 + i]; // K = +16..23
        // four B fragments (pre-packed, contiguous 32B per lane each)
        const __bf16* bb = Bp + ((((size_t)(cg * 4) * KB + kb) * 32 + lane) << 4);
        const size_t bstep = (size_t)KB << 9;       // KB*32*16 bf16 per col tile
        v16bf b0 = *(const v16bf*)(bb);
        v16bf b1 = *(const v16bf*)(bb + bstep);
        v16bf b2 = *(const v16bf*)(bb + 2 * bstep);
        v16bf b3 = *(const v16bf*)(bb + 3 * bstep);
        c0 = __builtin_amdgcn_wmma_f32_16x16x32_bf16(false, a, false, b0, (short)0, c0, false, false);
        c1 = __builtin_amdgcn_wmma_f32_16x16x32_bf16(false, a, false, b1, (short)0, c1, false, false);
        c2 = __builtin_amdgcn_wmma_f32_16x16x32_bf16(false, a, false, b2, (short)0, c2, false, false);
        c3 = __builtin_amdgcn_wmma_f32_16x16x32_bf16(false, a, false, b3, (short)0, c3, false, false);
    }
    if (A2) {
        const int KB2 = K2 >> 5;
        for (int kb = 0; kb < KB2; ++kb) {
            const float* ap = A2 + (size_t)arow * K2 + kb * 32 + koff;
            v16bf a;
#pragma unroll
            for (int i = 0; i < 8; ++i) a[i] = (__bf16)ap[i];
#pragma unroll
            for (int i = 0; i < 8; ++i) a[8 + i] = (__bf16)ap[16 + i];
            const __bf16* bb = B2p + ((((size_t)(cg * 4) * KB2 + kb) * 32 + lane) << 4);
            const size_t bstep = (size_t)KB2 << 9;
            v16bf b0 = *(const v16bf*)(bb);
            v16bf b1 = *(const v16bf*)(bb + bstep);
            v16bf b2 = *(const v16bf*)(bb + 2 * bstep);
            v16bf b3 = *(const v16bf*)(bb + 3 * bstep);
            c0 = __builtin_amdgcn_wmma_f32_16x16x32_bf16(false, a, false, b0, (short)0, c0, false, false);
            c1 = __builtin_amdgcn_wmma_f32_16x16x32_bf16(false, a, false, b1, (short)0, c1, false, false);
            c2 = __builtin_amdgcn_wmma_f32_16x16x32_bf16(false, a, false, b2, (short)0, c2, false, false);
            c3 = __builtin_amdgcn_wmma_f32_16x16x32_bf16(false, a, false, b3, (short)0, c3, false, false);
        }
    }

    // Epilogue. C/D layout (ISA 7.12.2): lanes 0-15 -> M=r, lanes 16-31 -> M=8+r.
    const int row0 = tm * 16 + ((lane & 16) ? 8 : 0);
    v8f acc[4] = {c0, c1, c2, c3};
#pragma unroll
    for (int t = 0; t < 4; ++t) {
        const int col = (cg * 4 + t) * 16 + (lane & 15);
        const float bv = bias  ? bias[col]  : 0.0f;
        const float rv = rsvec ? rsvec[col] : 0.0f;
#pragma unroll
        for (int r = 0; r < 8; ++r) {
            const int row = row0 + r;
            float v = acc[t][r] + bv;
            if (rowsc)     v += rowsc[row] * rv;
            if (relu_flag) v = v > 0.0f ? v : 0.0f;
            if (Cf) Cf[(size_t)row * Nout + col] = v;
            if (Cb) Cb[(size_t)row * Nout + col] = (__bf16)v;
        }
    }
}

// ---------------------------------------------------------------------------
// Edge kernel: one wave per edge.
//   t = relu( pre_src[src] + pre_dst[dst] + attr@W1_attr + b1 )   (128-wide)
//   agg[dst] += t   (fp32 atomics; W2/b2 are folded out at node level)
// ---------------------------------------------------------------------------
struct bh4 { __bf16 x, y, z, w; };

__global__ __launch_bounds__(256) void edge_msg(
    const int* __restrict__ ei, const float* __restrict__ eattr,
    const __bf16* __restrict__ psrc, const __bf16* __restrict__ pdst,
    const float* __restrict__ Wattr, const float* __restrict__ b1,
    float* __restrict__ agg, int e_)
{
    int e = blockIdx.x * 8 + (threadIdx.x >> 5);
    if (e >= e_) return;
    const int lane = threadIdx.x & 31;
    const int s = ei[e];
    const int d = ei[e_ + e];
    const int c0 = lane * 4;

    bh4 s4 = *(const bh4*)(psrc + (size_t)s * GH + c0);
    bh4 d4 = *(const bh4*)(pdst + (size_t)d * GH + c0);
    float t0 = (float)s4.x + (float)d4.x + b1[c0 + 0];
    float t1 = (float)s4.y + (float)d4.y + b1[c0 + 1];
    float t2 = (float)s4.z + (float)d4.z + b1[c0 + 2];
    float t3 = (float)s4.w + (float)d4.w + b1[c0 + 3];
#pragma unroll
    for (int j = 0; j < GED; ++j) {
        const float a = eattr[(size_t)e * GED + j];
        const float* wr = Wattr + j * GH + c0;
        t0 += a * wr[0]; t1 += a * wr[1]; t2 += a * wr[2]; t3 += a * wr[3];
    }
    float* ag = agg + (size_t)d * GH + c0;
    atomicAdd(&ag[0], t0 > 0.f ? t0 : 0.f);
    atomicAdd(&ag[1], t1 > 0.f ? t1 : 0.f);
    atomicAdd(&ag[2], t2 > 0.f ? t2 : 0.f);
    atomicAdd(&ag[3], t3 > 0.f ? t3 : 0.f);
}

// ---------------------------------------------------------------------------
// Mean pooling over batch segments.
// ---------------------------------------------------------------------------
__global__ void pool_sum(const float* __restrict__ ne, const int* __restrict__ batch,
                         float* __restrict__ sums, float* __restrict__ cnts) {
    int idx = blockIdx.x * 256 + threadIdx.x;
    if (idx >= GN * GDOUT) return;
    int node = idx / GDOUT, c = idx % GDOUT;
    int b = batch[node];
    atomicAdd(&sums[b * GDOUT + c], ne[idx]);
    if (c == 0) atomicAdd(&cnts[b], 1.0f);
}

__global__ void pool_div(const float* __restrict__ sums, const float* __restrict__ cnts,
                         float* __restrict__ out) {
    int idx = blockIdx.x * 256 + threadIdx.x;
    if (idx >= GB * GDOUT) return;
    float cc = cnts[idx / GDOUT];
    out[idx] = sums[idx] / (cc > 1.0f ? cc : 1.0f);
}

// ---------------------------------------------------------------------------
// Host-side launch.
// ---------------------------------------------------------------------------
extern "C" void kernel_launch(void* const* d_in, const int* in_sizes, int n_in,
                              void* d_out, int out_size, void* d_ws, size_t ws_size,
                              hipStream_t stream) {
    (void)in_sizes; (void)n_in; (void)out_size; (void)ws_size;

    const float* x       = (const float*)d_in[0];
    const int*   ei      = (const int*)  d_in[1];   // [2,E]
    const float* eattr   = (const float*)d_in[2];
    const int*   batch   = (const int*)  d_in[4];
    const float* emb_W   = (const float*)d_in[5];
    const float* emb_b   = (const float*)d_in[6];
    const float* nW1     = (const float*)d_in[7];
    const float* nb1     = (const float*)d_in[8];
    const float* nW2     = (const float*)d_in[9];
    const float* nb2     = (const float*)d_in[10];
    const float* eW1     = (const float*)d_in[11];  // [L,261,128]
    const float* eb1     = (const float*)d_in[12];
    const float* eW2     = (const float*)d_in[13];
    const float* eb2     = (const float*)d_in[14];
    const float* uW      = (const float*)d_in[15];  // [L,256,128]
    const float* ub      = (const float*)d_in[16];
    const float* oW      = (const float*)d_in[17];
    const float* ob      = (const float*)d_in[18];
    float* out = (float*)d_out;

    // bump allocator over d_ws
    char* wsp = (char*)d_ws;
    auto alloc = [&](size_t bytes) -> char* {
        char* p = wsp;
        wsp += (bytes + 255) & ~(size_t)255;
        return p;
    };
    float*  h    = (float*) alloc((size_t)GN * GH * 4);
    float*  t1   = (float*) alloc((size_t)GN * GH * 4);
    float*  hn   = (float*) alloc((size_t)GN * GH * 4);
    float*  agg  = (float*) alloc((size_t)GN * GH * 4);
    __bf16* psrc = (__bf16*)alloc((size_t)GN * GH * 2);
    __bf16* pdst = (__bf16*)alloc((size_t)GN * GH * 2);
    float*  deg  = (float*) alloc((size_t)GN * 4);
    float*  WcF  = (float*) alloc((size_t)GL * GH * GH * 4);
    float*  vc   = (float*) alloc((size_t)GL * GH * 4);
    float*  sums = (float*) alloc((size_t)GB * GDOUT * 4);
    float*  cnts = (float*) alloc((size_t)GB * 4);
    __bf16* Pemb = (__bf16*)alloc((size_t)GDIN * GH * 2);
    __bf16* PnW1 = (__bf16*)alloc((size_t)GL * GH * GH * 2);
    __bf16* PnW2 = (__bf16*)alloc((size_t)GL * GH * GH * 2);
    __bf16* Pes  = (__bf16*)alloc((size_t)GL * GH * GH * 2);
    __bf16* Ped  = (__bf16*)alloc((size_t)GL * GH * GH * 2);
    __bf16* Pwc  = (__bf16*)alloc((size_t)GL * GH * GH * 2);
    __bf16* Pup  = (__bf16*)alloc((size_t)GL * GH * GH * 2);
    __bf16* Pout = (__bf16*)alloc((size_t)GH * GDOUT * 2);

    auto zero = [&](float* p, int n) {
        zero_f32<<<(n + 255) / 256, 256, 0, stream>>>(p, n);
    };
    auto pack = [&](const float* W, __bf16* P, int K, int Nout, int ld) {
        pack_w<<<(K * Nout + 255) / 256, 256, 0, stream>>>(W, P, K, Nout, ld);
    };
    auto gemm = [&](const float* A, const __bf16* Bp, const float* A2,
                    const __bf16* B2p, const float* bias, const float* rsc,
                    const float* rsv, float* Cf, __bf16* Cb,
                    int M, int K, int K2, int Nout, int relu) {
        int tiles = (M >> 4) * (Nout >> 6);
        gemm_wmma<<<(tiles + 7) / 8, 256, 0, stream>>>(
            A, Bp, A2, B2p, bias, rsc, rsv, Cf, Cb, M, K, K2, Nout, relu);
    };

    const size_t HH = (size_t)GH * GH;             // 16384
    const size_t E1 = (size_t)(2 * GH + GED) * GH; // 261*128 rows of edge_W1

    // ---- per-launch precompute (deterministic) ----
    pack(emb_W, Pemb, GDIN, GH, GH);
    pack(oW, Pout, GH, GDOUT, GDOUT);
    for (int l = 0; l < GL; ++l) {
        pack(nW1 + l * HH, PnW1 + l * HH, GH, GH, GH);
        pack(nW2 + l * HH, PnW2 + l * HH, GH, GH, GH);
        pack(eW1 + l * E1,            Pes + l * HH, GH, GH, GH);  // rows 0..127  (src)
        pack(eW1 + l * E1 + GH * GH,  Ped + l * HH, GH, GH, GH);  // rows 128..255 (dst)
        pack(uW + l * 2 * HH,         Pup + l * HH, GH, GH, GH);  // updW_top
        // Wc = edge_W2 @ updW_bot ; vc = edge_b2 @ updW_bot
        mm128<<<64, 256, 0, stream>>>(eW2 + l * HH, uW + l * 2 * HH + HH, WcF + l * HH);
        vecmat128<<<1, 128, 0, stream>>>(eb2 + l * GH, uW + l * 2 * HH + HH, vc + l * GH);
        pack(WcF + l * HH, Pwc + l * HH, GH, GH, GH);
    }
    zero(deg, GN);
    degree_k<<<(GE + 255) / 256, 256, 0, stream>>>(ei + GE, deg, GE);

    // ---- embedding: h = x @ emb_W + emb_b ----
    gemm(x, Pemb, nullptr, nullptr, emb_b, nullptr, nullptr,
         h, nullptr, GN, GDIN, 0, GH, 0);

    // ---- layers ----
    for (int l = 0; l < GL; ++l) {
        // node MLP
        gemm(h, PnW1 + l * HH, nullptr, nullptr, nb1 + l * GH, nullptr, nullptr,
             t1, nullptr, GN, GH, 0, GH, 1);
        gemm(t1, PnW2 + l * HH, nullptr, nullptr, nb2 + l * GH, nullptr, nullptr,
             hn, nullptr, GN, GH, 0, GH, 0);
        // per-node pre-projections for the edge MLP (bf16 to halve gather bytes)
        gemm(h, Pes + l * HH, nullptr, nullptr, nullptr, nullptr, nullptr,
             nullptr, psrc, GN, GH, 0, GH, 0);
        gemm(h, Ped + l * HH, nullptr, nullptr, nullptr, nullptr, nullptr,
             nullptr, pdst, GN, GH, 0, GH, 0);
        // edge gather + relu + scatter-add (W2/b2 folded out by linearity)
        zero(agg, GN * GH);
        edge_msg<<<(GE + 7) / 8, 256, 0, stream>>>(
            ei, eattr, psrc, pdst,
            eW1 + l * E1 + 2 * HH,   // attr rows 256..260 of edge_W1
            eb1 + l * GH, agg, GE);
        // h' = hn@updW_top + agg@Wc + updb + deg (.) vc
        gemm(hn, Pup + l * HH, agg, Pwc + l * HH, ub + l * GH, deg, vc + l * GH,
             h, nullptr, GN, GH, GH, GH, 0);
    }

    // ---- output head + mean pooling ----
    gemm(h, Pout, nullptr, nullptr, ob, nullptr, nullptr,
         out, nullptr, GN, GH, 0, GDOUT, 0);
    zero(sums, GB * GDOUT);
    zero(cnts, GB);
    pool_sum<<<(GN * GDOUT + 255) / 256, 256, 0, stream>>>(out, batch, sums, cnts);
    pool_div<<<(GB * GDOUT + 255) / 256, 256, 0, stream>>>(
        sums, cnts, out + (size_t)GN * GDOUT);
}
